// SRMHA_78168404787419
// MI455X (gfx1250) — matrix operand order repaired and checked by reading, not compiled, not run on hardware
//
#include <hip/hip_runtime.h>
#include <hip/hip_bf16.h>

typedef __attribute__((ext_vector_type(16))) _Float16 v16h;
typedef __attribute__((ext_vector_type(8)))  _Float16 v8h;
typedef __attribute__((ext_vector_type(8)))  float    v8f;

namespace {
constexpr int CB  = 4;     // batch
constexpr int CN  = 4096;  // query tokens
constexpr int CC  = 384;   // channels
constexpr int CNH = 6;     // heads
constexpr int CHD = 64;    // head dim
constexpr int CNK = 1024;  // reduced (key) tokens = 32*32
constexpr int CW  = 64;    // spatial H = W = 64
}

// ---------------------------------------------------------------------------
// WMMA fragment loaders (wave32, v_wmma_f32_16x16x32_f16)
// A: 16(M) x 32(K) f16. Lane L holds row M=L%16.
//   halves 0..7  -> K = klo + 0..7,  halves 8..15 -> K = klo + 8..15,
//   klo = 16*(L/16). Two contiguous 16-byte loads per lane.
// ---------------------------------------------------------------------------
__device__ __forceinline__ v16h load_a16x32(const _Float16* A, int lda, int lane) {
  const int row = lane & 15;
  const int klo = (lane & 16);                // 0 or 16
  const _Float16* p = A + (size_t)row * lda + klo;
  v8h lo = *(const v8h*)(p);
  v8h hi = *(const v8h*)(p + 8);
  v16h a;
#pragma unroll
  for (int i = 0; i < 8; ++i) { a[i] = lo[i]; a[i + 8] = hi[i]; }
  return a;
}

// B: 32(K) x 16(N) f16 with B[k][n] = Wt[n][k] (Wt row-major over n, contiguous k).
// Lane L holds column N=L%16, halves h -> K = 16*(L/16) + h.
// One contiguous 32-byte chunk per lane.
__device__ __forceinline__ v16h load_bT32x16(const _Float16* Wt, int ldw, int lane) {
  const int n  = lane & 15;
  const int kb = lane & 16;                   // 0 or 16
  return *(const v16h*)(Wt + (size_t)n * ldw + kb);
}

// ---------------------------------------------------------------------------
// f32 -> f16 elementwise
// ---------------------------------------------------------------------------
__global__ void f32_to_f16_kernel(const float* __restrict__ in,
                                  _Float16* __restrict__ out, int n) {
  int i = blockIdx.x * blockDim.x + threadIdx.x;
  if (i < n) out[i] = (_Float16)in[i];
}

// ---------------------------------------------------------------------------
// Depthwise 2x2 stride-2 conv + LayerNorm over channels, f16 output.
// One block per (b, reduced-token), 384 threads (one per channel).
// ---------------------------------------------------------------------------
__global__ void __launch_bounds__(384)
convln_kernel(const float* __restrict__ x, const float* __restrict__ srw,
              const float* __restrict__ srb, const float* __restrict__ lng,
              const float* __restrict__ lnb, _Float16* __restrict__ xred) {
  const int bid = blockIdx.x;          // b*1024 + nk
  const int b  = bid >> 10;
  const int nk = bid & 1023;
  const int i = nk >> 5, j = nk & 31;
  const int c = threadIdx.x;
  const float* xb = x + (size_t)b * CN * CC;
  const int p00 = (2 * i) * CW + 2 * j;
  float v = xb[(size_t)p00 * CC + c]            * srw[c * 4 + 0]
          + xb[(size_t)(p00 + 1) * CC + c]      * srw[c * 4 + 1]
          + xb[(size_t)(p00 + CW) * CC + c]     * srw[c * 4 + 2]
          + xb[(size_t)(p00 + CW + 1) * CC + c] * srw[c * 4 + 3]
          + srb[c];
  __shared__ float red0[16], red1[16];
  float s = v, s2 = v * v;
#pragma unroll
  for (int m = 16; m >= 1; m >>= 1) { s += __shfl_xor(s, m); s2 += __shfl_xor(s2, m); }
  const int lane = threadIdx.x & 31, wid = threadIdx.x >> 5;
  if (lane == 0) { red0[wid] = s; red1[wid] = s2; }
  __syncthreads();
  if (wid == 0) {
    float a  = (lane < 12) ? red0[lane] : 0.0f;
    float b2 = (lane < 12) ? red1[lane] : 0.0f;
#pragma unroll
    for (int m = 8; m >= 1; m >>= 1) { a += __shfl_xor(a, m); b2 += __shfl_xor(b2, m); }
    if (lane == 0) { red0[0] = a; red1[0] = b2; }
  }
  __syncthreads();
  const float mean = red0[0] * (1.0f / CC);
  const float var  = red1[0] * (1.0f / CC) - mean * mean;
  const float inv  = rsqrtf(var + 1e-5f);
  xred[(size_t)bid * CC + c] = (_Float16)((v - mean) * inv * lng[c] + lnb[c]);
}

// ---------------------------------------------------------------------------
// WMMA GEMM:  out[M,N] = A[M,K](f16) @ W[N,K]^T (f16) + bias  (x @ W.T form)
// One wave per 16x64 output strip: the A fragment is loaded once per k-step
// and amortized over 4 WMMAs (4 adjacent N tiles). mode 0: f16, mode 1: f32.
// ---------------------------------------------------------------------------
__global__ void __launch_bounds__(256)
gemm_xwT_kernel(const _Float16* __restrict__ A, const _Float16* __restrict__ W,
                const float* __restrict__ bias, void* __restrict__ outp,
                int M, int N, int K, int mode) {
  const int lane = threadIdx.x & 31;
  const int wave = blockIdx.x * (blockDim.x >> 5) + (threadIdx.x >> 5);
  const int stripsN = N >> 6;                 // strips of 64 columns
  const int strips  = (M >> 4) * stripsN;
  if (wave >= strips) return;
  const int tm = wave / stripsN;
  const int tn = (wave - tm * stripsN) << 2;  // first 16-wide tile of the strip
  const _Float16* Ap = A + (size_t)tm * 16 * K;
  const _Float16* Wp0 = W + (size_t)(tn + 0) * 16 * K;
  const _Float16* Wp1 = W + (size_t)(tn + 1) * 16 * K;
  const _Float16* Wp2 = W + (size_t)(tn + 2) * 16 * K;
  const _Float16* Wp3 = W + (size_t)(tn + 3) * 16 * K;
  v8f c0 = {}, c1 = {}, c2 = {}, c3 = {};
  for (int k0 = 0; k0 < K; k0 += 32) {
    v16h a  = load_a16x32(Ap + k0, K, lane);
    v16h b0 = load_bT32x16(Wp0 + k0, K, lane);
    v16h b1 = load_bT32x16(Wp1 + k0, K, lane);
    v16h b2 = load_bT32x16(Wp2 + k0, K, lane);
    v16h b3 = load_bT32x16(Wp3 + k0, K, lane);
    c0 = __builtin_amdgcn_wmma_f32_16x16x32_f16(false, a, false, b0, (short)0, c0, false, false);
    c1 = __builtin_amdgcn_wmma_f32_16x16x32_f16(false, a, false, b1, (short)0, c1, false, false);
    c2 = __builtin_amdgcn_wmma_f32_16x16x32_f16(false, a, false, b2, (short)0, c2, false, false);
    c3 = __builtin_amdgcn_wmma_f32_16x16x32_f16(false, a, false, b3, (short)0, c3, false, false);
  }
  const int n  = lane & 15;
  const int rb = (lane & 16) >> 1;            // rows 0..7 or 8..15
  float bn0 = 0.0f, bn1 = 0.0f, bn2 = 0.0f, bn3 = 0.0f;
  if (bias) {
    bn0 = bias[(tn + 0) * 16 + n];
    bn1 = bias[(tn + 1) * 16 + n];
    bn2 = bias[(tn + 2) * 16 + n];
    bn3 = bias[(tn + 3) * 16 + n];
  }
  if (mode == 0) {
    _Float16* out = (_Float16*)outp;
#pragma unroll
    for (int r = 0; r < 8; ++r) {
      _Float16* o = out + (size_t)(tm * 16 + rb + r) * N + tn * 16 + n;
      o[0]  = (_Float16)(c0[r] + bn0);
      o[16] = (_Float16)(c1[r] + bn1);
      o[32] = (_Float16)(c2[r] + bn2);
      o[48] = (_Float16)(c3[r] + bn3);
    }
  } else {
    float* out = (float*)outp;
#pragma unroll
    for (int r = 0; r < 8; ++r) {
      float* o = out + (size_t)(tm * 16 + rb + r) * N + tn * 16 + n;
      o[0]  = c0[r] + bn0;
      o[16] = c1[r] + bn1;
      o[32] = c2[r] + bn2;
      o[48] = c3[r] + bn3;
    }
  }
}

// ---------------------------------------------------------------------------
// V (B*Nk, C) row-major -> Vt (B*C, Nk): per-head (d, key) layout so attention
// PV B-fragments are contiguous loads.
// ---------------------------------------------------------------------------
__global__ void transpose_v_kernel(const _Float16* __restrict__ v,
                                   _Float16* __restrict__ vt) {
  int idx = blockIdx.x * blockDim.x + threadIdx.x;
  if (idx >= CB * CNK * CC) return;
  int c   = idx % CC;
  int row = idx / CC;
  int b   = row >> 10;
  int key = row & 1023;
  vt[((size_t)b * CC + c) * CNK + key] = v[idx];
}

// ---------------------------------------------------------------------------
// Flash-style sign-softmax attention. One wave owns 16 q-rows of one (b,h).
// Streams keys in chunks of 32: 4 WMMAs for S, online |s|-softmax with sign
// folded into P, P staged via LDS into A-fragment layout, 4 WMMAs into O.
// ---------------------------------------------------------------------------
__global__ void __launch_bounds__(256)
attn_kernel(const _Float16* __restrict__ Q, const _Float16* __restrict__ Kf,
            const _Float16* __restrict__ Vt, _Float16* __restrict__ Ao) {
  const int lane  = threadIdx.x & 31;
  const int wslot = threadIdx.x >> 5;
  const int wave  = blockIdx.x * (blockDim.x >> 5) + wslot;
  const int NQT   = CN / 16;                  // 256 q-tiles per (b,h)
  const int bh    = wave / NQT;
  if (bh >= CB * CNH) return;
  const int qt = wave - bh * NQT;
  const int b  = bh / CNH, h = bh - b * CNH;

  const _Float16* qp = Q  + (size_t)b * CN * CC + (size_t)qt * 16 * CC + h * CHD;
  const _Float16* kp = Kf + (size_t)b * CNK * CC + h * CHD;
  const _Float16* vp = Vt + ((size_t)b * CC + h * CHD) * CNK;

  __shared__ __align__(32) _Float16 pbuf[8][16 * 32];
  _Float16* P = &pbuf[wslot][0];

  // Q fragments, reused for all key chunks (d = 0..31, 32..63)
  const v16h aq0 = load_a16x32(qp, CC, lane);
  const v16h aq1 = load_a16x32(qp + 32, CC, lane);

  v8f o0 = {}, o1 = {}, o2 = {}, o3 = {};
  float m[8], l[8];
#pragma unroll
  for (int r = 0; r < 8; ++r) { m[r] = 0.0f; l[r] = 0.0f; }  // |s| >= 0

  const float scale = 0.125f;                 // HD^-0.5

  for (int k0 = 0; k0 < CNK; k0 += 32) {
    // S[16q x 32key] in two 16x16 f32 fragments
    v8f s0 = {}, s1 = {};
    {
      v16h b0 = load_bT32x16(kp + (size_t)k0 * CC, CC, lane);
      v16h b1 = load_bT32x16(kp + (size_t)k0 * CC + 32, CC, lane);
      s0 = __builtin_amdgcn_wmma_f32_16x16x32_f16(false, aq0, false, b0, (short)0, s0, false, false);
      s0 = __builtin_amdgcn_wmma_f32_16x16x32_f16(false, aq1, false, b1, (short)0, s0, false, false);
    }
    {
      v16h b0 = load_bT32x16(kp + (size_t)(k0 + 16) * CC, CC, lane);
      v16h b1 = load_bT32x16(kp + (size_t)(k0 + 16) * CC + 32, CC, lane);
      s1 = __builtin_amdgcn_wmma_f32_16x16x32_f16(false, aq0, false, b0, (short)0, s1, false, false);
      s1 = __builtin_amdgcn_wmma_f32_16x16x32_f16(false, aq1, false, b1, (short)0, s1, false, false);
    }
    if (k0 + 32 < CNK) __builtin_prefetch(kp + (size_t)(k0 + 32) * CC, 0, 1);

    // sign / abs; per-row running max across the 32 keys.
    // C-layout: lane holds column n = lane%16 of rows r + 8*(lane/16), so a
    // row reduction is a 16-lane shfl_xor reduction within each half-wave.
    float as0[8], as1[8], sg0[8], sg1[8], tmax[8];
#pragma unroll
    for (int r = 0; r < 8; ++r) {
      float x0 = s0[r] * scale, x1 = s1[r] * scale;
      sg0[r] = (x0 < 0.0f) ? -1.0f : 1.0f;
      sg1[r] = (x1 < 0.0f) ? -1.0f : 1.0f;
      as0[r] = fabsf(x0); as1[r] = fabsf(x1);
      tmax[r] = fmaxf(as0[r], as1[r]);
    }
#pragma unroll
    for (int mk = 1; mk <= 8; mk <<= 1) {
#pragma unroll
      for (int r = 0; r < 8; ++r) tmax[r] = fmaxf(tmax[r], __shfl_xor(tmax[r], mk));
    }

    const int prow = (lane & 16) >> 1;
    const int pcol = lane & 15;
#pragma unroll
    for (int r = 0; r < 8; ++r) {
      float mn    = fmaxf(m[r], tmax[r]);
      float alpha = __expf(m[r] - mn);
      float e0    = __expf(as0[r] - mn);
      float e1    = __expf(as1[r] - mn);
      float rs    = e0 + e1;
#pragma unroll
      for (int mk = 1; mk <= 8; mk <<= 1) rs += __shfl_xor(rs, mk);
      l[r] = l[r] * alpha + rs;
      m[r] = mn;
      o0[r] *= alpha; o1[r] *= alpha; o2[r] *= alpha; o3[r] *= alpha;
      // P = sign * exp(|s| - m), staged to LDS row-major 16x32
      P[(prow + r) * 32 + pcol]      = (_Float16)(sg0[r] * e0);
      P[(prow + r) * 32 + 16 + pcol] = (_Float16)(sg1[r] * e1);
    }
    asm volatile("s_wait_dscnt 0x0" ::: "memory");

    // Reload P in A-fragment layout; PV: B[key][d] = Vt[d][key] (contiguous)
    v16h ap  = load_a16x32(P, 32, lane);
    v16h bv0 = load_bT32x16(vp + (size_t)0  * CNK + k0, CNK, lane);
    v16h bv1 = load_bT32x16(vp + (size_t)16 * CNK + k0, CNK, lane);
    v16h bv2 = load_bT32x16(vp + (size_t)32 * CNK + k0, CNK, lane);
    v16h bv3 = load_bT32x16(vp + (size_t)48 * CNK + k0, CNK, lane);
    o0 = __builtin_amdgcn_wmma_f32_16x16x32_f16(false, ap, false, bv0, (short)0, o0, false, false);
    o1 = __builtin_amdgcn_wmma_f32_16x16x32_f16(false, ap, false, bv1, (short)0, o1, false, false);
    o2 = __builtin_amdgcn_wmma_f32_16x16x32_f16(false, ap, false, bv2, (short)0, o2, false, false);
    o3 = __builtin_amdgcn_wmma_f32_16x16x32_f16(false, ap, false, bv3, (short)0, o3, false, false);
  }

  // Epilogue: divide by softmax denominator, store f16 (B*N, C) head slice.
  const int n  = lane & 15;
  const int rb = (lane & 16) >> 1;
  _Float16* op = Ao + ((size_t)b * CN + (size_t)qt * 16) * CC + h * CHD;
#pragma unroll
  for (int r = 0; r < 8; ++r) {
    float inv = 1.0f / l[r];
    op[(size_t)(rb + r) * CC + 0  + n] = (_Float16)(o0[r] * inv);
    op[(size_t)(rb + r) * CC + 16 + n] = (_Float16)(o1[r] * inv);
    op[(size_t)(rb + r) * CC + 32 + n] = (_Float16)(o2[r] * inv);
    op[(size_t)(rb + r) * CC + 48 + n] = (_Float16)(o3[r] * inv);
  }
}

// ---------------------------------------------------------------------------
// Host-side launch
// ---------------------------------------------------------------------------
extern "C" void kernel_launch(void* const* d_in, const int* in_sizes, int n_in,
                              void* d_out, int out_size, void* d_ws, size_t ws_size,
                              hipStream_t stream) {
  (void)in_sizes; (void)n_in; (void)out_size; (void)ws_size;
  const float* x   = (const float*)d_in[0];
  const float* Wq  = (const float*)d_in[1];
  const float* bq  = (const float*)d_in[2];
  const float* Wk  = (const float*)d_in[3];
  const float* bk  = (const float*)d_in[4];
  const float* Wv  = (const float*)d_in[5];
  const float* bv  = (const float*)d_in[6];
  const float* srw = (const float*)d_in[7];
  const float* srb = (const float*)d_in[8];
  const float* lng = (const float*)d_in[9];
  const float* lnb = (const float*)d_in[10];
  const float* Wp  = (const float*)d_in[11];
  const float* bp  = (const float*)d_in[12];
  // d_in[13]=H, d_in[14]=W are compile-time constants (64) in the reference.

  char* ws = (char*)d_ws;
  size_t off = 0;
  auto alloc = [&](size_t bytes) {
    char* p = ws + off;
    off = (off + bytes + 255) & ~(size_t)255;
    return p;
  };
  _Float16* xf    = (_Float16*)alloc((size_t)CB * CN * CC * 2);
  _Float16* wqf   = (_Float16*)alloc((size_t)CC * CC * 2);
  _Float16* wkf   = (_Float16*)alloc((size_t)CC * CC * 2);
  _Float16* wvf   = (_Float16*)alloc((size_t)CC * CC * 2);
  _Float16* wpf   = (_Float16*)alloc((size_t)CC * CC * 2);
  _Float16* xredf = (_Float16*)alloc((size_t)CB * CNK * CC * 2);
  _Float16* qf    = (_Float16*)alloc((size_t)CB * CN * CC * 2);
  _Float16* kf    = (_Float16*)alloc((size_t)CB * CNK * CC * 2);
  _Float16* vf    = (_Float16*)alloc((size_t)CB * CNK * CC * 2);
  _Float16* vtf   = (_Float16*)alloc((size_t)CB * CNK * CC * 2);
  _Float16* aof   = (_Float16*)alloc((size_t)CB * CN * CC * 2);

  // 1) f16 copies of x and weights
  {
    int nx = CB * CN * CC;
    f32_to_f16_kernel<<<(nx + 255) / 256, 256, 0, stream>>>(x, xf, nx);
    int nw = CC * CC;
    f32_to_f16_kernel<<<(nw + 255) / 256, 256, 0, stream>>>(Wq, wqf, nw);
    f32_to_f16_kernel<<<(nw + 255) / 256, 256, 0, stream>>>(Wk, wkf, nw);
    f32_to_f16_kernel<<<(nw + 255) / 256, 256, 0, stream>>>(Wv, wvf, nw);
    f32_to_f16_kernel<<<(nw + 255) / 256, 256, 0, stream>>>(Wp, wpf, nw);
  }
  // 2) spatial reduction conv + LayerNorm
  convln_kernel<<<CB * CNK, CC, 0, stream>>>(x, srw, srb, lng, lnb, xredf);
  // 3) projections (WMMA GEMMs, 16x64 strips per wave)
  {
    int strips = (CB * CN / 16) * (CC / 64);
    gemm_xwT_kernel<<<(strips + 7) / 8, 256, 0, stream>>>(xf, wqf, bq, qf,
                                                          CB * CN, CC, CC, 0);
  }
  {
    int strips = (CB * CNK / 16) * (CC / 64);
    gemm_xwT_kernel<<<(strips + 7) / 8, 256, 0, stream>>>(xredf, wkf, bk, kf,
                                                          CB * CNK, CC, CC, 0);
    gemm_xwT_kernel<<<(strips + 7) / 8, 256, 0, stream>>>(xredf, wvf, bv, vf,
                                                          CB * CNK, CC, CC, 0);
  }
  // 4) V -> per-head transposed layout
  {
    int nt = CB * CNK * CC;
    transpose_v_kernel<<<(nt + 255) / 256, 256, 0, stream>>>(vf, vtf);
  }
  // 5) flash sign-softmax attention
  {
    int waves = CB * CNH * (CN / 16);
    attn_kernel<<<(waves + 7) / 8, 256, 0, stream>>>(qf, kf, vtf, aof);
  }
  // 6) output projection (f32 out)
  {
    int strips = (CB * CN / 16) * (CC / 64);
    gemm_xwT_kernel<<<(strips + 7) / 8, 256, 0, stream>>>(aof, wpf, bp, d_out,
                                                          CB * CN, CC, CC, 1);
  }
}
